// LlamaAttention_82025285419201
// MI455X (gfx1250) — matrix operand order repaired and checked by reading, not compiled, not run on hardware
//
#include <hip/hip_runtime.h>
#include <hip/hip_bf16.h>

// ---------------------------------------------------------------------------
// Llama attention layer for MI455X (gfx1250), wave32 + WMMA bf16 + TDM.
// B=2, S=2048, H=4096, NH=32, NKV=8, HD=128, GQA groups = 4.
//
// - All matmuls: v_wmma_f32_16x16x32_bf16 (bf16 in, fp32 accumulate).
// - GEMMs: Tensor Data Mover (tensor_load_to_lds) streams A/B K-slices into
//   double-buffered LDS; waves compute from LDS while TDM fetches ahead
//   (TENSORcnt-synchronized, in-order per wave).
// - Attention: flash-style, scores never touch HBM; online softmax with
//   width-16 cross-lane reductions matching the wave32 WMMA C/D layout.
// ---------------------------------------------------------------------------

#define B_   2
#define S_   2048
#define H_   4096
#define NH_  32
#define NKV_ 8
#define HD_  128
#define GROUPS_ (NH_ / NKV_)

typedef __attribute__((ext_vector_type(16))) __bf16        v16bf;
typedef __attribute__((ext_vector_type(8)))  float         v8f;
typedef __attribute__((ext_vector_type(4)))  unsigned int  u32x4;
typedef __attribute__((ext_vector_type(8)))  int           i32x8;
typedef __attribute__((ext_vector_type(4)))  int           i32x4;

// Fragment: 16 bf16 values per lane = 8 VGPRs (A or B operand of 16x16x32).
union FragB { v16bf v; u32x4 q[2]; unsigned short u[16]; };

__device__ __forceinline__ unsigned short f2bf(float f) {
  unsigned u = __builtin_bit_cast(unsigned, f);
  u += 0x7FFFu + ((u >> 16) & 1u);         // round-to-nearest-even
  return (unsigned short)(u >> 16);
}

// Low 32 bits of a flat shared pointer == wave-relative LDS byte offset.
__device__ __forceinline__ unsigned lds_off(const void* p) {
  return (unsigned)(unsigned long long)(uintptr_t)p;
}

// ---------------------------------------------------------------------------
// TDM: DMA a 2-D tile (rows x 32 bf16) from a row-major [tensor_rows x K]
// bf16 matrix into LDS. Descriptor packing per CDNA5 ISA ch.8 (D# groups).
//   group0: count=1 | lds_addr | global_addr[56:0] | type=2
//   group1: data_size=2B, tensor_dim0=K, tensor_dim1=tensor_rows,
//           tile_dim0=32, tile_dim1=rows, tensor_dim0_stride=K
// Issued with inline asm (toolchain-portable); tracked by TENSORcnt.
// ---------------------------------------------------------------------------
__device__ __forceinline__ void tdm_load_tile32(
    unsigned ldsaddr, const unsigned short* gptr,
    unsigned rows, unsigned tensor_rows, unsigned K) {
  const unsigned long long ga = (unsigned long long)(uintptr_t)gptr;
  u32x4 g0;
  g0[0] = 1u;                                              // count=1, user D#
  g0[1] = ldsaddr;                                         // lds_addr (bytes)
  g0[2] = (unsigned)ga;                                    // global_addr[31:0]
  g0[3] = (unsigned)((ga >> 32) & 0x01FFFFFFull)           // global_addr[56:32]
        | (2u << 30);                                      // type = 2 (image)
  const unsigned td0 = K;                                  // tensor_dim0
  const unsigned td1 = tensor_rows;                        // tensor_dim1
  const unsigned long long s0 = (unsigned long long)K;     // dim0 stride
  i32x8 g1;
  g1[0] = (int)(1u << 16);                                 // data_size = 2B
  g1[1] = (int)((td0 & 0xFFFFu) << 16);                    // td0[15:0]
  g1[2] = (int)((td0 >> 16) | ((td1 & 0xFFFFu) << 16));    // td0 hi | td1 lo
  g1[3] = (int)((td1 >> 16) | (32u << 16));                // td1 hi | tile_dim0
  g1[4] = (int)rows;                                       // tile_dim1 (tile_dim2=0)
  g1[5] = (int)(unsigned)(s0 & 0xFFFFFFFFull);             // stride0[31:0]
  g1[6] = (int)(unsigned)(s0 >> 32);                       // stride0[47:32]
  g1[7] = 0;
  const i32x4 gz = {0, 0, 0, 0};
  asm volatile("tensor_load_to_lds %0, %1, %2, %3"
               :
               : "s"(g0), "s"(g1), "s"(gz), "s"(gz)
               : "memory");
}

__device__ __forceinline__ void wait_tensorcnt_le2() {
  asm volatile("s_wait_tensorcnt 0x2" ::: "memory");
}
__device__ __forceinline__ void wait_tensorcnt_0() {
  asm volatile("s_wait_tensorcnt 0x0" ::: "memory");
}

// ---------------------------------------------------------------------------
// fp32 -> bf16 elementwise conversion (grid-stride).
// ---------------------------------------------------------------------------
__global__ void cvt_f32_bf16(const float* __restrict__ in,
                             unsigned short* __restrict__ out, int n) {
  int i      = blockIdx.x * blockDim.x + threadIdx.x;
  int stride = gridDim.x * blockDim.x;
  for (; i < n; i += stride) out[i] = f2bf(in[i]);
}

// ---------------------------------------------------------------------------
// GEMM: C[m,n] = sum_k A[m,k] * Bw[n,k]  (A [M,K] bf16, Bw [N,K] bf16, C f32)
// One wave per block; 16x64 output tile. TDM streams the A (16x32) and
// B (64x32) K-slices into double-buffered LDS; WMMA consumes from LDS while
// the next slice is in flight (TENSORcnt <= 2 retires the older pair).
// Fragment layouts used here (wave32; A/B in contiguous k order, C/D per ISA 7.12.2):
//   A 16x32: lane l row=l%16; elems 0..15 -> K+(l/16)*16 (one 32B run).
//   B 32x16: lane l col=l%16; elems 0..15 -> K+(l/16)*16 (one 32B run).
//   C/D:     lane l, vgpr v -> row = v + 8*(l/16), col = l%16.
// ---------------------------------------------------------------------------
__global__ __launch_bounds__(32) void gemm_bf16_abt(
    const unsigned short* __restrict__ A,
    const unsigned short* __restrict__ Bw,
    float* __restrict__ C, int K, int N, int Mtot) {
  __shared__ unsigned short lA[2][16 * 32];   // 2 x 1 KB
  __shared__ unsigned short lB[2][64 * 32];   // 2 x 4 KB
  const int lane = threadIdx.x;
  const int half = lane >> 4, lr = lane & 15;
  const int m0 = blockIdx.y * 16;
  const int n0 = blockIdx.x * 64;

  v8f acc[4];
#pragma unroll
  for (int t = 0; t < 4; ++t)
#pragma unroll
    for (int i = 0; i < 8; ++i) acc[t][i] = 0.0f;

  // Prologue: fetch K-slice 0 into buffer 0.
  tdm_load_tile32(lds_off(&lA[0][0]), A + (size_t)m0 * K, 16, (unsigned)Mtot,
                  (unsigned)K);
  tdm_load_tile32(lds_off(&lB[0][0]), Bw + (size_t)n0 * K, 64, (unsigned)N,
                  (unsigned)K);

  for (int k = 0; k < K; k += 32) {
    const int cur = (k >> 5) & 1;
    if (k + 32 < K) {
      const int nxt = cur ^ 1;
      tdm_load_tile32(lds_off(&lA[nxt][0]), A + (size_t)m0 * K + k + 32, 16,
                      (unsigned)Mtot, (unsigned)K);
      tdm_load_tile32(lds_off(&lB[nxt][0]), Bw + (size_t)n0 * K + k + 32, 64,
                      (unsigned)N, (unsigned)K);
      wait_tensorcnt_le2();   // older pair (cur) has landed; newer 2 in flight
    } else {
      wait_tensorcnt_0();
    }

    FragB a;
    a.q[0] = *(const u32x4*)(&lA[cur][lr * 32 + half * 16]);
    a.q[1] = *(const u32x4*)(&lA[cur][lr * 32 + half * 16 + 8]);
#pragma unroll
    for (int t = 0; t < 4; ++t) {
      FragB b;
      b.q[0] = *(const u32x4*)(&lB[cur][(t * 16 + lr) * 32 + half * 16]);
      b.q[1] = *(const u32x4*)(&lB[cur][(t * 16 + lr) * 32 + half * 16 + 8]);
      acc[t] = __builtin_amdgcn_wmma_f32_16x16x32_bf16(
          false, a.v, false, b.v, (short)0, acc[t], false, false);
    }
  }

#pragma unroll
  for (int t = 0; t < 4; ++t)
#pragma unroll
    for (int v = 0; v < 8; ++v)
      C[(size_t)(m0 + v + 8 * half) * N + n0 + t * 16 + lr] = acc[t][v];
}

// ---------------------------------------------------------------------------
// RoPE (fp32 math) + relayout to head-major bf16 [B, nheads, S, HD].
// ---------------------------------------------------------------------------
__global__ void rope_kernel(const float* __restrict__ src,
                            const int* __restrict__ pos_ids,
                            unsigned short* __restrict__ dst, int nheads) {
  const int total = B_ * S_ * nheads * (HD_ / 2);
  int idx = blockIdx.x * blockDim.x + threadIdx.x;
  if (idx >= total) return;
  const int i = idx & 63;
  const int h = (idx >> 6) % nheads;
  const int s = (idx / (64 * nheads)) % S_;
  const int b = idx / (64 * nheads * S_);

  const float pos  = (float)pos_ids[b * S_ + s];
  const float freq = pos * __powf(10000.0f, -(float)i * (1.0f / 64.0f));
  const float c = __cosf(freq), sn = __sinf(freq);

  const float* row = src + ((size_t)(b * S_ + s)) * (nheads * HD_) + h * HD_;
  const float x1 = row[i], x2 = row[i + 64];
  const size_t ob = (((size_t)b * nheads + h) * S_ + s) * HD_;
  dst[ob + i]      = f2bf(x1 * c - x2 * sn);
  dst[ob + i + 64] = f2bf(x2 * c + x1 * sn);
}

// V: fp32 [B,S,NKV*HD] -> bf16 transposed [B, NKV, HD, S]  (keys contiguous,
// so the P*V B-fragment is a contiguous 32-byte load per lane).
__global__ void v_transpose(const float* __restrict__ src,
                            unsigned short* __restrict__ dst) {
  const int total = B_ * S_ * NKV_ * HD_;
  int idx = blockIdx.x * blockDim.x + threadIdx.x;
  if (idx >= total) return;
  const int d = idx & 127;
  const int h = (idx >> 7) % NKV_;
  const int s = (idx / (128 * NKV_)) % S_;
  const int b = idx / (128 * NKV_ * S_);
  dst[(((size_t)b * NKV_ + h) * HD_ + d) * S_ + s] =
      f2bf(src[((size_t)b * S_ + s) * (NKV_ * HD_) + h * HD_ + d]);
}

// ---------------------------------------------------------------------------
// Flash attention. One wave per (b, h, 16-row q-block).
// Q strip 16xHD lives in 4 A-frags; key blocks of 32; online softmax.
// P (D-layout) is staged through LDS and reloaded in A-layout for P*V.
// K/V blocks are shared by 4 q-heads and all q-blocks -> L2-resident (192MB).
// ---------------------------------------------------------------------------
__global__ __launch_bounds__(32) void flash_attn(
    const unsigned short* __restrict__ Qr,   // [B,NH,S,HD]
    const unsigned short* __restrict__ Kr,   // [B,NKV,S,HD]
    const unsigned short* __restrict__ Vt,   // [B,NKV,HD,S]
    unsigned short* __restrict__ Obf) {      // [B,S,NH*HD]
  __shared__ unsigned short ldsP[16 * 32];

  const int lane = threadIdx.x, half = lane >> 4, lr = lane & 15;
  const int qb = blockIdx.x, h = blockIdx.y, b = blockIdx.z;
  const int kvh = h / GROUPS_;
  const int q0 = qb * 16;
  const float scale = 0.08838834764831845f;   // 1/sqrt(128)

  // Q A-fragments for the whole head dim (4 x 16x32).
  const unsigned short* qbase =
      Qr + (((size_t)b * NH_ + h) * S_ + (q0 + lr)) * HD_;
  FragB aq[4];
#pragma unroll
  for (int kk = 0; kk < 4; ++kk) {
    aq[kk].q[0] = *(const u32x4*)(qbase + kk * 32 + half * 16);
    aq[kk].q[1] = *(const u32x4*)(qbase + kk * 32 + half * 16 + 8);
  }

  v8f o[8];
  float mrow[8], rsum[8];
#pragma unroll
  for (int t = 0; t < 8; ++t)
#pragma unroll
    for (int v = 0; v < 8; ++v) o[t][v] = 0.0f;
#pragma unroll
  for (int v = 0; v < 8; ++v) { mrow[v] = -1e30f; rsum[v] = 0.0f; }

  const unsigned short* kbase = Kr + ((size_t)b * NKV_ + kvh) * S_ * HD_;
  const unsigned short* vbase = Vt + ((size_t)b * NKV_ + kvh) * HD_ * S_;

  for (int kb = 0; kb <= q0 + 15; kb += 32) {
    // ---- scores: 2 key tiles x 4 K-steps of 32 over HD ----
    v8f sc[2];
#pragma unroll
    for (int nt = 0; nt < 2; ++nt) {
      v8f acc;
#pragma unroll
      for (int i = 0; i < 8; ++i) acc[i] = 0.0f;
      const unsigned short* krow =
          kbase + (size_t)(kb + nt * 16 + lr) * HD_ + half * 16;
#pragma unroll
      for (int kk = 0; kk < 4; ++kk) {
        FragB bk;
        bk.q[0] = *(const u32x4*)(krow + kk * 32);
        bk.q[1] = *(const u32x4*)(krow + kk * 32 + 8);
        acc = __builtin_amdgcn_wmma_f32_16x16x32_bf16(
            false, aq[kk].v, false, bk.v, (short)0, acc, false, false);
      }
      sc[nt] = acc;
    }

    // ---- scale + causal mask + per-row block max ----
    float mblk[8];
#pragma unroll
    for (int v = 0; v < 8; ++v) {
      const int qr = q0 + v + 8 * half;
      float s0 = sc[0][v] * scale; if (kb + lr > qr)      s0 = -1e30f;
      float s1 = sc[1][v] * scale; if (kb + 16 + lr > qr) s1 = -1e30f;
      sc[0][v] = s0; sc[1][v] = s1;
      mblk[v] = fmaxf(s0, s1);
    }
#pragma unroll
    for (int v = 0; v < 8; ++v)
      for (int mk = 1; mk < 16; mk <<= 1)
        mblk[v] = fmaxf(mblk[v], __shfl_xor(mblk[v], mk, 32));

    float alpha[8], mnew[8], ls[8];
#pragma unroll
    for (int v = 0; v < 8; ++v) {
      mnew[v]  = fmaxf(mrow[v], mblk[v]);
      alpha[v] = __expf(mrow[v] - mnew[v]);   // first block: exp(-huge)=0
      mrow[v]  = mnew[v];
      ls[v]    = 0.0f;
    }

    // ---- exp, lane-partial row sums, stage P (bf16) into LDS ----
#pragma unroll
    for (int v = 0; v < 8; ++v) {
      const int row = v + 8 * half;
#pragma unroll
      for (int nt = 0; nt < 2; ++nt) {
        const float p = __expf(sc[nt][v] - mnew[v]);
        ls[v] += p;
        ldsP[row * 32 + nt * 16 + lr] = f2bf(p);
      }
    }
#pragma unroll
    for (int v = 0; v < 8; ++v) {
      for (int mk = 1; mk < 16; mk <<= 1)
        ls[v] += __shfl_xor(ls[v], mk, 32);
      rsum[v] = rsum[v] * alpha[v] + ls[v];
    }
#pragma unroll
    for (int t = 0; t < 8; ++t)
#pragma unroll
      for (int v = 0; v < 8; ++v) o[t][v] *= alpha[v];

    __syncthreads();

    // ---- reload P in A-layout, then P*V into O accumulators ----
    FragB pf;
    pf.q[0] = *(const u32x4*)(&ldsP[lr * 32 + half * 16]);
    pf.q[1] = *(const u32x4*)(&ldsP[lr * 32 + half * 16 + 8]);
#pragma unroll
    for (int t = 0; t < 8; ++t) {
      const unsigned short* vrow =
          vbase + (size_t)(t * 16 + lr) * S_ + kb + half * 16;
      FragB vf;
      vf.q[0] = *(const u32x4*)(vrow);
      vf.q[1] = *(const u32x4*)(vrow + 8);
      o[t] = __builtin_amdgcn_wmma_f32_16x16x32_bf16(
          false, pf.v, false, vf.v, (short)0, o[t], false, false);
    }
    __syncthreads();
  }

  // ---- normalize and store bf16 attn output [B,S,NH*HD] ----
#pragma unroll
  for (int t = 0; t < 8; ++t)
#pragma unroll
    for (int v = 0; v < 8; ++v) {
      const float val = o[t][v] / rsum[v];
      Obf[((size_t)b * S_ + q0 + v + 8 * half) * (NH_ * HD_) + h * HD_ +
          t * 16 + lr] = f2bf(val);
    }
}

// ---------------------------------------------------------------------------
// Host-side launcher.
// Inputs: 0=hidden [B,S,H] f32, 1=position_ids [B,S] i32,
//         2=wq [4096,4096], 3=wk [1024,4096], 4=wv [1024,4096],
//         5=wo [4096,4096]  (all f32). Output: [B,S,H] f32.
// ---------------------------------------------------------------------------
extern "C" void kernel_launch(void* const* d_in, const int* in_sizes, int n_in,
                              void* d_out, int out_size, void* d_ws,
                              size_t ws_size, hipStream_t stream) {
  (void)in_sizes; (void)n_in; (void)out_size; (void)ws_size;

  const float* hidden = (const float*)d_in[0];
  const int*   posids = (const int*)d_in[1];
  const float* wq     = (const float*)d_in[2];
  const float* wk     = (const float*)d_in[3];
  const float* wv     = (const float*)d_in[4];
  const float* wo     = (const float*)d_in[5];
  float*       out    = (float*)d_out;

  const size_t M  = (size_t)B_ * S_;   // 4096 token rows
  const size_t NQ = (size_t)NH_ * HD_; // 4096
  const size_t NK = (size_t)NKV_ * HD_;// 1024

  char* ws = (char*)d_ws;
  size_t off = 0;
  auto take = [&](size_t bytes) {
    char* p = ws + off;
    off = (off + bytes + 255) & ~(size_t)255;
    return p;
  };
  unsigned short* hid_bf = (unsigned short*)take(M * H_ * 2);
  unsigned short* wq_bf  = (unsigned short*)take(NQ * H_ * 2);
  unsigned short* wk_bf  = (unsigned short*)take(NK * H_ * 2);
  unsigned short* wv_bf  = (unsigned short*)take(NK * H_ * 2);
  unsigned short* wo_bf  = (unsigned short*)take((size_t)H_ * NQ * 2);
  float*          Qf     = (float*)take(M * NQ * 4);
  float*          Kf     = (float*)take(M * NK * 4);
  float*          Vf     = (float*)take(M * NK * 4);
  unsigned short* Qr     = (unsigned short*)take(M * NQ * 2);
  unsigned short* Kr     = (unsigned short*)take(M * NK * 2);
  unsigned short* Vt     = (unsigned short*)take(M * NK * 2);
  unsigned short* Abf    = (unsigned short*)take(M * NQ * 2);

  // 1) fp32 -> bf16 conversions
  const int CT = 256, CG = 4096;
  cvt_f32_bf16<<<CG, CT, 0, stream>>>(hidden, hid_bf, (int)(M * H_));
  cvt_f32_bf16<<<CG, CT, 0, stream>>>(wq, wq_bf, (int)(NQ * H_));
  cvt_f32_bf16<<<CG, CT, 0, stream>>>(wk, wk_bf, (int)(NK * H_));
  cvt_f32_bf16<<<CG, CT, 0, stream>>>(wv, wv_bf, (int)(NK * H_));
  cvt_f32_bf16<<<CG, CT, 0, stream>>>(wo, wo_bf, (int)(H_ * NQ));

  // 2) QKV projections (A * B^T, TDM-fed bf16 WMMA, fp32 out)
  gemm_bf16_abt<<<dim3((unsigned)(NQ / 64), (unsigned)(M / 16)), 32, 0,
                  stream>>>(hid_bf, wq_bf, Qf, H_, (int)NQ, (int)M);
  gemm_bf16_abt<<<dim3((unsigned)(NK / 64), (unsigned)(M / 16)), 32, 0,
                  stream>>>(hid_bf, wk_bf, Kf, H_, (int)NK, (int)M);
  gemm_bf16_abt<<<dim3((unsigned)(NK / 64), (unsigned)(M / 16)), 32, 0,
                  stream>>>(hid_bf, wv_bf, Vf, H_, (int)NK, (int)M);

  // 3) RoPE (fp32) + relayout to bf16; V transposed for P*V fragments
  {
    int tq = B_ * S_ * NH_ * (HD_ / 2);
    rope_kernel<<<(tq + 255) / 256, 256, 0, stream>>>(Qf, posids, Qr, NH_);
    int tk = B_ * S_ * NKV_ * (HD_ / 2);
    rope_kernel<<<(tk + 255) / 256, 256, 0, stream>>>(Kf, posids, Kr, NKV_);
    int tv = B_ * S_ * NKV_ * HD_;
    v_transpose<<<(tv + 255) / 256, 256, 0, stream>>>(Vf, Vt);
  }

  // 4) Flash attention (one wave per 16 q-rows per head)
  flash_attn<<<dim3(S_ / 16, NH_, B_), 32, 0, stream>>>(Qr, Kr, Vt, Abf);

  // 5) Output projection -> fp32 result
  gemm_bf16_abt<<<dim3((unsigned)(H_ / 64), (unsigned)(M / 16)), 32, 0,
                  stream>>>(Abf, wo_bf, out, (int)NQ, H_, (int)M);
}
